// Enc_coor_47914655154488
// MI455X (gfx1250) — matrix operand order repaired and checked by reading, not compiled, not run on hardware
//
#include <hip/hip_runtime.h>
#include <math.h>

typedef _Float16 f16;
typedef _Float16 v16h __attribute__((ext_vector_type(16)));
typedef float    v8f  __attribute__((ext_vector_type(8)));

#define SS   16
#define BB   16
#define KK   3
#define TT   20
#define FPX  64
#define DPX  28
#define CPX  37
#define PP   1369
#define PPAD 1376
#define HHH  128
#define FSTR 80            // padded LDS frame row stride (halves); cols 64..79 = 0

union Frag {
    struct { uint4 a, b; } q;   // two 16-byte chunks -> 8 VGPRs
    v16h v;
};

static __device__ __forceinline__ v8f wmma_f16(v16h a, v16h b, v8f c) {
    // 8 args: (neg_a, A, neg_b, B, c_mod, C, reuse_a, reuse_b)
    return __builtin_amdgcn_wmma_f32_16x16x32_f16(false, a, false, b, (short)0, c,
                                                  false, false);
}

// ---------------------------------------------------------------------------
// Weight prep: w [K x N] f32 row-major  ->  out [N x Kp] f16, zero-padded K.
// ---------------------------------------------------------------------------
__global__ void wprep_kernel(const float* __restrict__ w, f16* __restrict__ out,
                             int K, int N, int Kp) {
    size_t total = (size_t)N * Kp;
    for (size_t i = (size_t)blockIdx.x * blockDim.x + threadIdx.x; i < total;
         i += (size_t)gridDim.x * blockDim.x) {
        int kk = (int)(i % Kp);
        int n  = (int)(i / Kp);
        out[i] = (kk < K) ? (f16)w[(size_t)kk * N + n] : (f16)0.0f;
    }
}

// digit (S,B,K,28,28) f32 -> tmpl[b][sk=s*3+k][dy][32] f16 (dx padded to 32)
__global__ void tprep_kernel(const float* __restrict__ digit, f16* __restrict__ tmpl) {
    size_t total = (size_t)BB * 48 * DPX * 32;
    for (size_t i = (size_t)blockIdx.x * blockDim.x + threadIdx.x; i < total;
         i += (size_t)gridDim.x * blockDim.x) {
        int dx = (int)(i % 32); size_t q = i / 32;
        int dy = (int)(q % DPX); q /= DPX;
        int sk = (int)(q % 48);
        int b  = (int)(q / 48);
        int s = sk / 3, k = sk % 3;
        tmpl[i] = (dx < DPX)
            ? (f16)digit[((((size_t)s * BB + b) * KK + k) * DPX + dy) * DPX + dx]
            : (f16)0.0f;
    }
}

// ---------------------------------------------------------------------------
// Correlation: one workgroup per (b,t).  LDS: f16 frame rows padded to 80 cols
// (pad cols zeroed -> branch-free B gather) + 48 zero-padded templates.
// Each wave computes 16x16 tiles (M = s*k rows, N = pixels) with
// v_wmma_f32_16x16x32_f16, K chunked per template row (dy), dx padded to 32.
// Two independent accumulator chains (even/odd dy) let the scheduler overlap
// ds_load_b128 of one chain with the WMMA of the other.
// ---------------------------------------------------------------------------
__global__ void conv_kernel(const float* __restrict__ frames,
                            const f16* __restrict__ tmpl,
                            float* __restrict__ conv) {
    extern __shared__ __attribute__((aligned(16))) char smem[];
    f16* sFrame = (f16*)smem;                  // 64*80 halves = 10 KB
    f16* sTmpl  = sFrame + FPX * FSTR;         // 48*28*32 halves = 84 KB

    const int bt = blockIdx.x;
    const int b = bt / TT, t = bt % TT;
    const int tid = threadIdx.x;

    // frame -> f16 LDS, rows padded to FSTR with zeros
    const float* fr = frames + ((size_t)b * TT + t) * (FPX * FPX);
    for (int i = tid; i < FPX * FSTR; i += 256) {
        int row = i / FSTR, col = i % FSTR;
        sFrame[i] = (col < FPX) ? (f16)fr[row * FPX + col] : (f16)0.0f;
    }

    const uint4* tsrc = (const uint4*)(tmpl + (size_t)b * 48 * DPX * 32);
    uint4* tdst = (uint4*)sTmpl;
    for (int i = tid; i < 48 * DPX * 32 / 8; i += 256) tdst[i] = tsrc[i];
    __syncthreads();

    const int wave = tid >> 5;
    const int lane = tid & 31;
    const int l16  = lane & 15;
    const int hi   = lane >> 4;
    const int klo  = hi * 8;

    for (int tile = wave; tile < 3 * 86; tile += 8) {   // 48/16 x ceil(1369/16)
        const int mt = tile / 86, nt = tile % 86;
        const int p  = nt * 16 + l16;
        const int pc = (p < PP) ? p : (PP - 1);
        const int py = pc / CPX, px = pc % CPX;
        const int arow = mt * 16 + l16;
        const uint4* ab = (const uint4*)(sTmpl + arow * (DPX * 32));
        const f16* fbase = sFrame + py * FSTR + px + klo;   // max col 67 < 80

        v8f c0 = {}, c1 = {};
        for (int dy = 0; dy < DPX; dy += 2) {
            // even chain
            {
                Frag A;
                A.q.a = ab[dy * 4 + hi];
                A.q.b = ab[dy * 4 + 2 + hi];
                const f16* frow = fbase + dy * FSTR;
                v16h bv;
#pragma unroll
                for (int e = 0; e < 16; ++e) bv[e] = frow[e < 8 ? e : e + 8];
                c0 = wmma_f16(A.v, bv, c0);
            }
            // odd chain (independent registers/accumulator)
            {
                Frag A;
                A.q.a = ab[(dy + 1) * 4 + hi];
                A.q.b = ab[(dy + 1) * 4 + 2 + hi];
                const f16* frow = fbase + (dy + 1) * FSTR;
                v16h bv;
#pragma unroll
                for (int e = 0; e < 16; ++e) bv[e] = frow[e < 8 ? e : e + 8];
                c1 = wmma_f16(A.v, bv, c1);
            }
        }
        const v8f c = c0 + c1;

        if (p < PP) {
#pragma unroll
            for (int r = 0; r < 8; ++r) {
                int sk = mt * 16 + r + hi * 8;        // D layout: M=r / M=8+r
                int s = sk / 3, k = sk % 3;
                size_t row = (((size_t)s * BB + b) * TT + t) * KK + k;
                conv[row * PP + p] = c[r];
            }
        }
    }
}

// ---------------------------------------------------------------------------
// Row softmax over P=1369, emit normalized f16 rows padded to 1376.
// ---------------------------------------------------------------------------
__global__ void softmax_kernel(const float* __restrict__ conv, f16* __restrict__ out) {
    __shared__ float red[256];
    const size_t row = blockIdx.x;
    const int tid = threadIdx.x;
    const float* x = conv + row * PP;

    float v[6];
    float mx = -3.4e38f;
#pragma unroll
    for (int j = 0; j < 6; ++j) {
        int i = tid + j * 256;
        v[j] = (i < PP) ? x[i] : -3.4e38f;
        mx = fmaxf(mx, v[j]);
    }
    red[tid] = mx; __syncthreads();
    for (int s = 128; s > 0; s >>= 1) {
        if (tid < s) red[tid] = fmaxf(red[tid], red[tid + s]);
        __syncthreads();
    }
    mx = red[0]; __syncthreads();

    float sum = 0.0f;
#pragma unroll
    for (int j = 0; j < 6; ++j) { v[j] = __expf(v[j] - mx); sum += v[j]; }
    red[tid] = sum; __syncthreads();
    for (int s = 128; s > 0; s >>= 1) {
        if (tid < s) red[tid] += red[tid + s];
        __syncthreads();
    }
    const float inv = 1.0f / red[0];

    f16* o = out + row * PPAD;
#pragma unroll
    for (int j = 0; j < 6; ++j) {
        int i = tid + j * 256;
        if (i < PPAD) o[i] = (f16)(v[j] * inv);   // pad cols get exp(-inf)=0
    }
}

// conv_disp = conv[t+1]-conv[t], f16 rows padded to 1376
__global__ void diff_kernel(const float* __restrict__ conv, f16* __restrict__ out,
                            size_t total) {
    for (size_t i = (size_t)blockIdx.x * blockDim.x + threadIdx.x; i < total;
         i += (size_t)gridDim.x * blockDim.x) {
        size_t r = i / PPAD;
        int cix = (int)(i % PPAD);
        if (cix >= PP) { out[i] = (f16)0.0f; continue; }
        int k = (int)(r % 3); size_t q = r / 3;
        int tt = (int)(q % (TT - 1)); q /= (TT - 1);
        int b = (int)(q % BB);
        int s = (int)(q / BB);
        size_t base = ((size_t)s * BB + b) * TT;
        size_t r1 = (base + tt + 1) * KK + k;
        size_t r0 = (base + tt) * KK + k;
        out[i] = (f16)(conv[r1 * PP + cix] - conv[r0 * PP + cix]);
    }
}

// ---------------------------------------------------------------------------
// GEMM + bias + ReLU, f16 in / f16 out, N-blocked by 4 (wave -> 16x64 strip).
// A: [M x Kp] f16 row-major (Kp mult of 32).  BT: [N x Kp] f16 (weightsT).
// Block = 8 waves; wave w -> mtile = blockIdx.x*8+w, ntiles = blockIdx.y*4..+3.
// All dims multiples of tile sizes -> EXEC stays all-ones around WMMA.
// ---------------------------------------------------------------------------
__global__ void gemm_kernel(const f16* __restrict__ A, const f16* __restrict__ BT,
                            const float* __restrict__ bias, f16* __restrict__ C,
                            int Kp, int Nstride) {
    const int lane = threadIdx.x & 31;
    const int wave = threadIdx.x >> 5;
    const int l16  = lane & 15;
    const int hi   = lane >> 4;
    const int mt  = blockIdx.x * 8 + wave;
    const int nt0 = blockIdx.y * 4;

    const uint4* aP = (const uint4*)(A + (size_t)(mt * 16 + l16) * Kp);
    const uint4* bP0 = (const uint4*)(BT + (size_t)((nt0 + 0) * 16 + l16) * Kp);
    const uint4* bP1 = (const uint4*)(BT + (size_t)((nt0 + 1) * 16 + l16) * Kp);
    const uint4* bP2 = (const uint4*)(BT + (size_t)((nt0 + 2) * 16 + l16) * Kp);
    const uint4* bP3 = (const uint4*)(BT + (size_t)((nt0 + 3) * 16 + l16) * Kp);

    v8f c0 = {}, c1 = {}, c2 = {}, c3 = {};
    const int ksteps = Kp >> 5;
#pragma unroll 2
    for (int kk = 0; kk < ksteps; ++kk) {
        Frag a;
        a.q.a = aP[kk * 4 + hi * 2];      // A: K {0..7 | 16..23}
        a.q.b = aP[kk * 4 + hi * 2 + 1];  // A: K {8..15 | 24..31}
        Frag b;
        b.q.a = bP0[kk * 4 + hi * 2];     // B: K {0..15 | 16..31}
        b.q.b = bP0[kk * 4 + hi * 2 + 1];
        c0 = wmma_f16(a.v, b.v, c0);
        b.q.a = bP1[kk * 4 + hi * 2];
        b.q.b = bP1[kk * 4 + hi * 2 + 1];
        c1 = wmma_f16(a.v, b.v, c1);
        b.q.a = bP2[kk * 4 + hi * 2];
        b.q.b = bP2[kk * 4 + hi * 2 + 1];
        c2 = wmma_f16(a.v, b.v, c2);
        b.q.a = bP3[kk * 4 + hi * 2];
        b.q.b = bP3[kk * 4 + hi * 2 + 1];
        c3 = wmma_f16(a.v, b.v, c3);
    }

    v8f acc[4] = { c0, c1, c2, c3 };
#pragma unroll
    for (int j = 0; j < 4; ++j) {
        const int ncol = (nt0 + j) * 16 + l16;
        const float bv = bias[ncol];
#pragma unroll
        for (int r = 0; r < 8; ++r) {
            float val = acc[j][r] + bv;
            val = val > 0.0f ? val : 0.0f;
            C[(size_t)(mt * 16 + r + hi * 8) * Nstride + ncol] = (f16)val;
        }
    }
}

// Final 128 -> 2 head: act=0 tanh, act=1 exp.  Output f32.
__global__ void head_kernel(const f16* __restrict__ X, const float* __restrict__ w,
                            const float* __restrict__ bsc, float* __restrict__ out,
                            int M, int act) {
    int r = blockIdx.x * blockDim.x + threadIdx.x;
    if (r >= M) return;
    const f16* x = X + (size_t)r * HHH;
    float a0 = 0.0f, a1 = 0.0f;
#pragma unroll 4
    for (int k = 0; k < HHH; ++k) {
        float xv = (float)x[k];
        a0 += xv * w[k * 2];
        a1 += xv * w[k * 2 + 1];
    }
    a0 += bsc[0]; a1 += bsc[1];
    if (act == 0) { a0 = tanhf(a0); a1 = tanhf(a1); }
    else          { a0 = __expf(a0); a1 = __expf(a1); }
    out[(size_t)r * 2]     = a0;
    out[(size_t)r * 2 + 1] = a1;
}

// ---------------------------------------------------------------------------
extern "C" void kernel_launch(void* const* d_in, const int* in_sizes, int n_in,
                              void* d_out, int out_size, void* d_ws, size_t ws_size,
                              hipStream_t stream) {
    const float* frames = (const float*)d_in[0];
    const float* digit  = (const float*)d_in[1];
    const float* w_h1   = (const float*)d_in[2];
    const float* b_h1   = (const float*)d_in[3];
    const float* wm1    = (const float*)d_in[4];
    const float* bm1    = (const float*)d_in[5];
    const float* wm2    = (const float*)d_in[6];
    const float* bm2    = (const float*)d_in[7];
    const float* ws1    = (const float*)d_in[8];
    const float* bs1    = (const float*)d_in[9];
    const float* ws2    = (const float*)d_in[10];
    const float* bs2    = (const float*)d_in[11];
    const float* wd1    = (const float*)d_in[12];
    const float* bd1    = (const float*)d_in[13];
    const float* wd2    = (const float*)d_in[14];
    const float* bd2    = (const float*)d_in[15];
    const float* wd3    = (const float*)d_in[16];
    const float* bd3    = (const float*)d_in[17];

    const size_t ROWS  = (size_t)SS * BB * TT * KK;        // 15360
    const size_t ROWSD = (size_t)SS * BB * (TT - 1) * KK;  // 14592

    char* ws = (char*)d_ws;
    size_t off = 0;
    auto alloc = [&](size_t bytes) -> char* {
        char* p = ws + off;
        off += (bytes + 255) & ~(size_t)255;
        return p;
    };
    float* conv   = (float*)alloc(ROWS * PP * 4);
    f16*   soft   = (f16*)alloc(ROWS * PPAD * 2);
    f16*   dispin = (f16*)alloc(ROWSD * PPAD * 2);
    f16*   hidden = (f16*)alloc(ROWS * 256 * 2);
    f16*   h1m    = (f16*)alloc(ROWS * 128 * 2);
    f16*   h1s    = (f16*)alloc(ROWS * 128 * 2);
    f16*   d1     = (f16*)alloc(ROWSD * 256 * 2);
    f16*   d2     = (f16*)alloc(ROWSD * 128 * 2);
    f16*   wh1T   = (f16*)alloc((size_t)256 * PPAD * 2);
    f16*   wd1T   = (f16*)alloc((size_t)256 * PPAD * 2);
    f16*   wm1T   = (f16*)alloc((size_t)128 * 256 * 2);
    f16*   ws1T   = (f16*)alloc((size_t)128 * 256 * 2);
    f16*   wd2T   = (f16*)alloc((size_t)128 * 256 * 2);
    f16*   tmpl   = (f16*)alloc((size_t)BB * 48 * DPX * 32 * 2);

    // weight / template prep
    wprep_kernel<<<1376, 256, 0, stream>>>(w_h1, wh1T, PP, 256, PPAD);
    wprep_kernel<<<1376, 256, 0, stream>>>(wd1,  wd1T, PP, 256, PPAD);
    wprep_kernel<<<128,  256, 0, stream>>>(wm1,  wm1T, 256, 128, 256);
    wprep_kernel<<<128,  256, 0, stream>>>(ws1,  ws1T, 256, 128, 256);
    wprep_kernel<<<128,  256, 0, stream>>>(wd2,  wd2T, 256, 128, 256);
    tprep_kernel<<<2688, 256, 0, stream>>>(digit, tmpl);

    // correlation: 320 workgroups; dyn LDS = padded frame + templates
    conv_kernel<<<BB * TT, 256, (FPX * FSTR + 48 * DPX * 32) * 2, stream>>>(
        frames, tmpl, conv);

    // softmax rows + temporal diff rows (f16, padded)
    softmax_kernel<<<(int)ROWS, 256, 0, stream>>>(conv, soft);
    {
        size_t total = ROWSD * PPAD;
        diff_kernel<<<(int)((total + 255) / 256), 256, 0, stream>>>(conv, dispin, total);
    }

    // WMMA GEMM chain (grid.y = Ntiles/4)
    gemm_kernel<<<dim3(120, 4), 256, 0, stream>>>(soft,   wh1T, b_h1, hidden, PPAD, 256);
    gemm_kernel<<<dim3(120, 2), 256, 0, stream>>>(hidden, wm1T, bm1,  h1m,    256,  128);
    gemm_kernel<<<dim3(120, 2), 256, 0, stream>>>(hidden, ws1T, bs1,  h1s,    256,  128);
    gemm_kernel<<<dim3(114, 4), 256, 0, stream>>>(dispin, wd1T, bd1,  d1,     PPAD, 256);
    gemm_kernel<<<dim3(114, 2), 256, 0, stream>>>(d1,     wd2T, bd2,  d2,     256,  128);

    // heads -> d_out (q_mean | q_std | disp)
    float* out = (float*)d_out;
    head_kernel<<<60, 256, 0, stream>>>(h1m, wm2, bm2, out,            (int)ROWS,  0);
    head_kernel<<<60, 256, 0, stream>>>(h1s, ws2, bs2, out + ROWS * 2, (int)ROWS,  1);
    head_kernel<<<57, 256, 0, stream>>>(d2,  wd3, bd3, out + ROWS * 4, (int)ROWSD, 0);
}
